// CryptEAGLE_17875653886366
// MI455X (gfx1250) — matrix operand order repaired and checked by reading, not compiled, not run on hardware
//
#include <hip/hip_runtime.h>
#include <math.h>

// ---------------------------------------------------------------------------
// CryptEAGLE graph-attention layer for MI455X (gfx1250, wave32, WMMA + TDM).
// * All GEMMs: v_wmma_f32_16x16x32_bf16 (f32 accumulate).
// * Weights pre-packed once to bf16 in exact B-fragment layout (L2-resident).
// * Uniform 2D fp32 tiles (edge_attr/x/agg) staged to LDS by the Tensor Data
//   Mover (tensor_load_to_lds, TENSORcnt), overlapping the row gathers.
// * e = edge_attr@WE^T recomputed in both edge passes (saves ~900MB HBM).
// ---------------------------------------------------------------------------

#define DEV __device__ __forceinline__

constexpr int N_NODES = 50000;
constexpr int N_EDGES = 600000;
constexpr int DIM     = 128;
constexpr int H       = 8;
constexpr int HD      = 16;      // one head == one 16-wide WMMA column tile
constexpr int ROWS_E  = 32;      // edges per block (2 WMMA row tiles)
constexpr int W_ELEMS = 8 * 4 * 32 * 16;  // packed bf16 elements per matrix

typedef __bf16 bf16;
typedef __attribute__((ext_vector_type(16))) __bf16        v16bf;
typedef __attribute__((ext_vector_type(8)))  float         v8f;
typedef __attribute__((ext_vector_type(4)))  unsigned int  u32x4;
typedef __attribute__((ext_vector_type(8)))  unsigned int  u32x8;

DEV v8f wmma_bf16(v16bf a, v16bf b, v8f c) {
  return __builtin_amdgcn_wmma_f32_16x16x32_bf16(false, a, false, b,
                                                 (short)0, c, false, false);
}

// ----- TDM: DMA a [rows x 128] fp32 tile (row-major, contiguous) to LDS ----
// D# group0: count=1 | lds_addr | global_addr | type=2 (ISA 8.3)
// D# group1: data_size=4B, tensor_dim0=128, tensor_dim1=rows, tile 128 x rows,
//            tensor_dim0_stride=128 (ISA 8.4). Groups 2/3 disabled (2D).
DEV void tdm_issue_tile_f32(const float* __restrict__ gsrc, void* lds_dst,
                            int rows) {
  const unsigned long long ga = (unsigned long long)(const void*)gsrc;
  const unsigned int ldsoff = (unsigned int)(unsigned long long)(size_t)lds_dst;
  const u32x4 g0 = { 1u,                                   // count=1
                     ldsoff,                               // lds_addr
                     (unsigned int)ga,                     // global_addr lo
                     ((unsigned int)(ga >> 32) & 0x01FFFFFFu) | (2u << 30) };
  const u32x8 g1 = { 2u << 16,                             // data_size=4B
                     ((unsigned)DIM) << 16,                // tensor_dim0=128
                     ((unsigned)rows) << 16,               // tensor_dim1
                     ((unsigned)DIM) << 16,                // tile_dim0=128
                     (unsigned)rows,                       // tile_dim1
                     (unsigned)DIM,                        // dim0_stride=128
                     0u, 0u };
  asm volatile("tensor_load_to_lds %0, %1" :: "s"(g0), "s"(g1) : "memory");
}

// ----- A fragment: 16x32 bf16 tile from LDS fp32 row-major [16][128] -------
// ISA 7.12.2: lane L holds row M=L&15; lanes>=16 take K-offset +8;
// element t -> K = 16*(t>>3) + 8*(L>>4) + (t&7) within the 32-wide k-step.
DEV v16bf a_frag_lds_f32(const float* rows, int kstep, int lane) {
  v16bf a;
  const int m  = lane & 15;
  const int kb = ((lane >> 4) & 1) * 8;
  const float* r = rows + m * DIM + kstep * 32;
#pragma unroll
  for (int t = 0; t < 16; ++t)
    a[t] = (bf16)r[((t >> 3) << 4) + kb + (t & 7)];
  return a;
}

// ----- B fragment from pre-packed bf16 weights: one 32B contiguous load ----
DEV v16bf b_frag_pk(const bf16* __restrict__ Wpk, int colTile, int kstep,
                    int lane) {
  return *(const v16bf*)(Wpk + ((((colTile << 2) | kstep) << 5) + lane) * 16);
}

// Butterfly sum over a 16-lane half (C/D rows live within one half).
DEV float half_reduce16(float v) {
#pragma unroll
  for (int m = 1; m <= 8; m <<= 1) v += __shfl_xor(v, m, 32);
  return v;
}

// ---------------------------------------------------------------------------
// Pack W (fp32 [128][128], used as x@W^T) into bf16 B-fragment order:
// packed[mtx][colTile][kstep][lane][t],  B[k][n] = W[n][k].
__global__ void k_pack_w(const float* __restrict__ W0, const float* __restrict__ W1,
                         const float* __restrict__ W2, const float* __restrict__ W3,
                         const float* __restrict__ W4, bf16* __restrict__ packed) {
  const int b = blockIdx.x;             // mtx*32 + colTile*4 + kstep
  const int mtx = b >> 5, colTile = (b >> 2) & 7, kstep = b & 3;
  const int lane = threadIdx.x;
  const float* W = (mtx == 0) ? W0 : (mtx == 1) ? W1 : (mtx == 2) ? W2
                                   : (mtx == 3) ? W3 : W4;
  const int n  = colTile * 16 + (lane & 15);
  const int k0 = kstep * 32 + ((lane >> 4) & 1) * 8;
  const float* src = W + n * DIM + k0;
  bf16* dst = packed + ((size_t)b * 32 + lane) * 16;
#pragma unroll
  for (int t = 0; t < 16; ++t) dst[t] = (bf16)src[((t >> 3) << 4) + (t & 7)];
}

// ---------------------------------------------------------------------------
// p = prototype @ Wp^T  (tiny GEMV)
__global__ void k_proto(const float* __restrict__ proto,
                        const float* __restrict__ Wp, float* __restrict__ p) {
  const int o = threadIdx.x;
  float s = 0.f;
  for (int d = 0; d < DIM; ++d) s += proto[d] * Wp[o * DIM + d];
  p[o] = s;
}

// ---------------------------------------------------------------------------
// q,k,v projections + prototype guidance fused. 1 block = 16 nodes, 8 waves;
// wave w owns head/col-tile w for all three matrices (12 WMMAs/wave).
__global__ __launch_bounds__(256) void k_proj_qkv(
    const float* __restrict__ x, const bf16* __restrict__ Wpk,  // QKV packed
    const float* __restrict__ p,
    float* __restrict__ q, float* __restrict__ k, float* __restrict__ v) {
  __shared__ float lds_x[16 * DIM];
  const int tid = threadIdx.x, lane = tid & 31, wave = tid >> 5;
  const size_t row0 = (size_t)blockIdx.x * 16;

  if (wave == 0) {
    tdm_issue_tile_f32(x + row0 * DIM, lds_x, 16);
    __builtin_amdgcn_s_wait_tensorcnt(0);
  }
  __syncthreads();

  v16bf a0 = a_frag_lds_f32(lds_x, 0, lane);
  v16bf a1 = a_frag_lds_f32(lds_x, 1, lane);
  v16bf a2 = a_frag_lds_f32(lds_x, 2, lane);
  v16bf a3 = a_frag_lds_f32(lds_x, 3, lane);

  const int n = lane & 15, hi = (lane >> 4) & 1;
  const float pn = p[wave * HD + n];
  float* Os[3] = {q, k, v};
#pragma unroll
  for (int mtx = 0; mtx < 3; ++mtx) {
    const bf16* W = Wpk + (size_t)mtx * W_ELEMS;
    v8f acc = {};
    acc = wmma_bf16(a0, b_frag_pk(W, wave, 0, lane), acc);
    acc = wmma_bf16(a1, b_frag_pk(W, wave, 1, lane), acc);
    acc = wmma_bf16(a2, b_frag_pk(W, wave, 2, lane), acc);
    acc = wmma_bf16(a3, b_frag_pk(W, wave, 3, lane), acc);
    float* O = Os[mtx];
#pragma unroll
    for (int r = 0; r < 8; ++r) {
      const int m = r + hi * 8;
      float val = acc[r];
      if (mtx == 0) {                    // q: align = relu(<q,p>_head)
        const float align = fmaxf(half_reduce16(val * pn), 0.f);
        val += 0.1f * align * pn;
      }
      O[(row0 + m) * DIM + wave * HD + n] = val;
    }
  }
}

// ---------------------------------------------------------------------------
// Edge pass 1: e = edge_attr@WE^T (WMMA), score = relu(<q_i, k_j+e>/4),
// scatter-add into row_sum. 1 block = 32 edges (2 row tiles, B-frags reused).
__global__ __launch_bounds__(256) void k_edge_score(
    const float* __restrict__ edge_attr, const int* __restrict__ eidx,
    const bf16* __restrict__ WEpk,
    const float* __restrict__ q, const float* __restrict__ k,
    float* __restrict__ score, float* __restrict__ row_sum) {
  __shared__ float lds_ea[ROWS_E * DIM];
  __shared__ float lds_q[ROWS_E * DIM];
  __shared__ float lds_k[ROWS_E * DIM];
  __shared__ int   s_src[ROWS_E], s_dst[ROWS_E];

  const int tid = threadIdx.x, lane = tid & 31, wave = tid >> 5;
  const int ebase = blockIdx.x * ROWS_E;

  if (wave == 0)                               // TDM overlaps the gathers
    tdm_issue_tile_f32(edge_attr + (size_t)ebase * DIM, lds_ea, ROWS_E);
  if (tid < ROWS_E) {
    s_src[tid] = eidx[ebase + tid];
    s_dst[tid] = eidx[N_EDGES + ebase + tid];
  }
  __syncthreads();                             // indices visible
  for (int i = tid; i < ROWS_E * DIM; i += 256) {  // coalesced row gathers
    const int rr = i >> 7, cc = i & 127;
    lds_k[i] = k[(size_t)s_src[rr] * DIM + cc];
    lds_q[i] = q[(size_t)s_dst[rr] * DIM + cc];
  }
  if (wave == 0) __builtin_amdgcn_s_wait_tensorcnt(0);
  __syncthreads();

  const v16bf b0 = b_frag_pk(WEpk, wave, 0, lane);
  const v16bf b1 = b_frag_pk(WEpk, wave, 1, lane);
  const v16bf b2 = b_frag_pk(WEpk, wave, 2, lane);
  const v16bf b3 = b_frag_pk(WEpk, wave, 3, lane);
  const int n = lane & 15, hi = (lane >> 4) & 1;
#pragma unroll
  for (int rt = 0; rt < 2; ++rt) {
    const float* ea = lds_ea + rt * 16 * DIM;
    v8f acc = {};
    acc = wmma_bf16(a_frag_lds_f32(ea, 0, lane), b0, acc);
    acc = wmma_bf16(a_frag_lds_f32(ea, 1, lane), b1, acc);
    acc = wmma_bf16(a_frag_lds_f32(ea, 2, lane), b2, acc);
    acc = wmma_bf16(a_frag_lds_f32(ea, 3, lane), b3, acc);
#pragma unroll
    for (int r = 0; r < 8; ++r) {
      const int m = rt * 16 + r + hi * 8;
      const int base = m * DIM + wave * HD + n;
      float t = lds_q[base] * (lds_k[base] + acc[r]);
      t = half_reduce16(t);                    // sum over head dim
      const float s = fmaxf(t * 0.25f, 0.f);   // /sqrt(16), relu
      if (n == 0) {
        score[(size_t)(ebase + m) * H + wave] = s;
        atomicAdd(&row_sum[(size_t)s_dst[m] * H + wave], s);  // L2-resident
      }
    }
  }
}

// ---------------------------------------------------------------------------
// Edge pass 2: recompute e, alpha = score/(row_sum+1e-6),
// agg[dst] += alpha*(v_j + e) via f32 atomics (agg fits in L2).
__global__ __launch_bounds__(256) void k_edge_msg(
    const float* __restrict__ edge_attr, const int* __restrict__ eidx,
    const bf16* __restrict__ WEpk, const float* __restrict__ v,
    const float* __restrict__ score, const float* __restrict__ row_sum,
    float* __restrict__ agg) {
  __shared__ float lds_ea[ROWS_E * DIM];
  __shared__ float lds_v[ROWS_E * DIM];
  __shared__ float lds_al[ROWS_E * H];
  __shared__ int   s_src[ROWS_E], s_dst[ROWS_E];

  const int tid = threadIdx.x, lane = tid & 31, wave = tid >> 5;
  const int ebase = blockIdx.x * ROWS_E;

  if (wave == 0)
    tdm_issue_tile_f32(edge_attr + (size_t)ebase * DIM, lds_ea, ROWS_E);
  if (tid < ROWS_E) {
    s_src[tid] = eidx[ebase + tid];
    s_dst[tid] = eidx[N_EDGES + ebase + tid];
  }
  __syncthreads();
  {                                            // alpha, all 256 threads
    const int m = tid >> 3, hh = tid & 7;
    const float s  = score[(size_t)(ebase + m) * H + hh];
    const float rs = row_sum[(size_t)s_dst[m] * H + hh];
    lds_al[tid] = s / (rs + 1e-6f);
  }
  for (int i = tid; i < ROWS_E * DIM; i += 256) {
    const int rr = i >> 7, cc = i & 127;
    lds_v[i] = v[(size_t)s_src[rr] * DIM + cc];
  }
  if (wave == 0) __builtin_amdgcn_s_wait_tensorcnt(0);
  __syncthreads();

  const v16bf b0 = b_frag_pk(WEpk, wave, 0, lane);
  const v16bf b1 = b_frag_pk(WEpk, wave, 1, lane);
  const v16bf b2 = b_frag_pk(WEpk, wave, 2, lane);
  const v16bf b3 = b_frag_pk(WEpk, wave, 3, lane);
  const int n = lane & 15, hi = (lane >> 4) & 1;
#pragma unroll
  for (int rt = 0; rt < 2; ++rt) {
    const float* ea = lds_ea + rt * 16 * DIM;
    v8f acc = {};
    acc = wmma_bf16(a_frag_lds_f32(ea, 0, lane), b0, acc);
    acc = wmma_bf16(a_frag_lds_f32(ea, 1, lane), b1, acc);
    acc = wmma_bf16(a_frag_lds_f32(ea, 2, lane), b2, acc);
    acc = wmma_bf16(a_frag_lds_f32(ea, 3, lane), b3, acc);
#pragma unroll
    for (int r = 0; r < 8; ++r) {
      const int m = rt * 16 + r + hi * 8;
      const float val = lds_al[m * H + wave] *
                        (lds_v[m * DIM + wave * HD + n] + acc[r]);
      atomicAdd(&agg[(size_t)s_dst[m] * DIM + wave * HD + n], val);
    }
  }
}

// ---------------------------------------------------------------------------
// out_pre = agg @ Wo^T + bo + x   (residual fused)
__global__ __launch_bounds__(256) void k_out_proj(
    const float* __restrict__ agg, const bf16* __restrict__ Wopk,
    const float* __restrict__ bo, const float* __restrict__ x,
    float* __restrict__ out_pre) {
  __shared__ float lds_a[16 * DIM];
  const int tid = threadIdx.x, lane = tid & 31, wave = tid >> 5;
  const size_t row0 = (size_t)blockIdx.x * 16;

  if (wave == 0) {
    tdm_issue_tile_f32(agg + row0 * DIM, lds_a, 16);
    __builtin_amdgcn_s_wait_tensorcnt(0);
  }
  __syncthreads();

  v8f acc = {};
  acc = wmma_bf16(a_frag_lds_f32(lds_a, 0, lane), b_frag_pk(Wopk, wave, 0, lane), acc);
  acc = wmma_bf16(a_frag_lds_f32(lds_a, 1, lane), b_frag_pk(Wopk, wave, 1, lane), acc);
  acc = wmma_bf16(a_frag_lds_f32(lds_a, 2, lane), b_frag_pk(Wopk, wave, 2, lane), acc);
  acc = wmma_bf16(a_frag_lds_f32(lds_a, 3, lane), b_frag_pk(Wopk, wave, 3, lane), acc);

  const int n = lane & 15, hi = (lane >> 4) & 1;
  const int col = wave * HD + n;
  const float bias = bo[col];
#pragma unroll
  for (int r = 0; r < 8; ++r) {
    const size_t idx = (row0 + r + hi * 8) * DIM + col;
    out_pre[idx] = acc[r] + bias + x[idx];
  }
}

// ---------------------------------------------------------------------------
// GraphNorm stats + finalize.
__global__ __launch_bounds__(256) void k_colsum(
    const float* __restrict__ out_pre, float* __restrict__ sums) {
  __shared__ float buf[256];
  const int col = threadIdx.x & 127, part = threadIdx.x >> 7;
  float s = 0.f;
  for (int row = blockIdx.x * 2 + part; row < N_NODES; row += gridDim.x * 2)
    s += out_pre[(size_t)row * DIM + col];
  buf[threadIdx.x] = s;
  __syncthreads();
  if (threadIdx.x < 128)
    atomicAdd(&sums[col], buf[threadIdx.x] + buf[threadIdx.x + 128]);
}

__global__ __launch_bounds__(256) void k_colvar(
    const float* __restrict__ out_pre, const float* __restrict__ sums,
    const float* __restrict__ gms, float* __restrict__ vars) {
  __shared__ float buf[256];
  const int col = threadIdx.x & 127, part = threadIdx.x >> 7;
  const float sub = gms[col] * (sums[col] * (1.f / N_NODES));
  float s = 0.f;
  for (int row = blockIdx.x * 2 + part; row < N_NODES; row += gridDim.x * 2) {
    const float c = out_pre[(size_t)row * DIM + col] - sub;
    s += c * c;
  }
  buf[threadIdx.x] = s;
  __syncthreads();
  if (threadIdx.x < 128)
    atomicAdd(&vars[col], buf[threadIdx.x] + buf[threadIdx.x + 128]);
}

__global__ __launch_bounds__(256) void k_finalize(
    const float* __restrict__ out_pre, const float* __restrict__ sums,
    const float* __restrict__ vars, const float* __restrict__ gw,
    const float* __restrict__ gb, const float* __restrict__ gms,
    float* __restrict__ out) {
  const size_t i = (size_t)blockIdx.x * 256 + threadIdx.x;
  if (i >= (size_t)N_NODES * DIM) return;
  const int col = (int)(i & 127);
  const float mean = sums[col] * (1.f / N_NODES);
  const float var  = vars[col] * (1.f / N_NODES);
  const float cen  = out_pre[i] - gms[col] * mean;
  out[i] = fmaxf(gw[col] * cen * rsqrtf(var + 1e-5f) + gb[col], 0.f);
}

// ---------------------------------------------------------------------------
extern "C" void kernel_launch(void* const* d_in, const int* in_sizes, int n_in,
                              void* d_out, int out_size, void* d_ws, size_t ws_size,
                              hipStream_t stream) {
  (void)in_sizes; (void)n_in; (void)out_size; (void)ws_size;
  const float* x    = (const float*)d_in[0];
  const float* eatt = (const float*)d_in[1];
  const float* prot = (const float*)d_in[2];
  const float* WQ   = (const float*)d_in[3];
  const float* WK   = (const float*)d_in[4];
  const float* WV   = (const float*)d_in[5];
  const float* WE   = (const float*)d_in[6];
  const float* Wp   = (const float*)d_in[7];
  const float* Wo   = (const float*)d_in[8];
  const float* bo   = (const float*)d_in[9];
  const float* gnw  = (const float*)d_in[10];
  const float* gnb  = (const float*)d_in[11];
  const float* gms  = (const float*)d_in[12];
  const int*   eidx = (const int*)d_in[13];
  float* out = (float*)d_out;

  char* ws = (char*)d_ws;
  const size_t NB = (size_t)N_NODES * DIM * sizeof(float);      // 25.6 MB
  const size_t SC = (size_t)N_EDGES * H * sizeof(float);        // 19.2 MB
  const size_t RS = (size_t)N_NODES * H * sizeof(float);        //  1.6 MB
  float* q       = (float*)(ws);
  float* k       = (float*)(ws + NB);
  float* v       = (float*)(ws + 2 * NB);
  float* agg     = (float*)(ws + 3 * NB);
  float* out_pre = (float*)(ws + 4 * NB);
  float* score   = (float*)(ws + 5 * NB);
  float* row_sum = (float*)(ws + 5 * NB + SC);
  float* pvec    = (float*)(ws + 5 * NB + SC + RS);
  float* sums    = pvec + DIM;
  float* vars    = sums + DIM;
  bf16*  packed  = (bf16*)(ws + 5 * NB + SC + RS + 1536);  // 5 x 32KB, 32B-aligned
  // packed order: [WQ, WK, WV, WE, Wo]
  bf16* WEpk = packed + (size_t)3 * W_ELEMS;
  bf16* Wopk = packed + (size_t)4 * W_ELEMS;

  hipMemsetAsync(agg, 0, NB, stream);
  hipMemsetAsync(row_sum, 0, RS, stream);
  hipMemsetAsync(sums, 0, DIM * sizeof(float), stream);
  hipMemsetAsync(vars, 0, DIM * sizeof(float), stream);

  k_pack_w    <<<5 * 32, 32, 0, stream>>>(WQ, WK, WV, WE, Wo, packed);
  k_proto     <<<1, DIM, 0, stream>>>(prot, Wp, pvec);
  k_proj_qkv  <<<N_NODES / 16, 256, 0, stream>>>(x, packed, pvec, q, k, v);
  k_edge_score<<<N_EDGES / ROWS_E, 256, 0, stream>>>(eatt, eidx, WEpk, q, k,
                                                     score, row_sum);
  k_edge_msg  <<<N_EDGES / ROWS_E, 256, 0, stream>>>(eatt, eidx, WEpk, v,
                                                     score, row_sum, agg);
  k_out_proj  <<<N_NODES / 16, 256, 0, stream>>>(agg, Wopk, bo, x, out_pre);
  k_colsum    <<<200, 256, 0, stream>>>(out_pre, sums);
  k_colvar    <<<200, 256, 0, stream>>>(out_pre, sums, gms, vars);
  k_finalize  <<<(N_NODES * DIM + 255) / 256, 256, 0, stream>>>(
      out_pre, sums, vars, gnw, gnb, gms, out);
}
